// STx_encoder_79731772883228
// MI455X (gfx1250) — matrix operand order repaired and checked by reading, not compiled, not run on hardware
//
#include <hip/hip_runtime.h>
#include <hip/hip_bf16.h>

typedef __attribute__((ext_vector_type(16))) __bf16 v16bf;
typedef __attribute__((ext_vector_type(8)))  __bf16 v8bf;
typedef __attribute__((ext_vector_type(8)))  float  v8f;

#define B_GR   8
#define NG_    4096
#define IN_CH_ 32
#define HID_   64
#define MP_    100
#define NTOT   (B_GR * NG_)      // 32768
#define NPOOL  (B_GR * MP_)      // 800
#define E_PPI_ 600

// ---------------------------------------------------------------------------
// Branchless WMMA tile loaders (bf16 row-major source).
// A: lane l -> row (l&15); K = hi*16+e, hi=l>>4
// B(from rows of X, i.e. B = X^T): lane l -> col (l&15) = row of X; K = hi*16+e
// FULL=true: no guards at all (aligned case). FULL=false: clamp row + select 0.
// ---------------------------------------------------------------------------
template<bool FULL>
__device__ inline v16bf load_A16(const __bf16* __restrict__ X, int ld, long rowBase,
                                 int nRows, int k0) {
  int lane = threadIdx.x & 31, r = lane & 15, hi = lane >> 4;
  int rc = FULL ? r : (r < nRows ? r : 0);
  const __bf16* p = X + (rowBase + rc) * (long)ld + k0 + hi * 16;
  v8bf lo = *(const v8bf*)p;
  v8bf hh = *(const v8bf*)(p + 8);
  v16bf a;
  bool ok = FULL || (r < nRows);
#pragma unroll
  for (int e = 0; e < 8; ++e) {
    a[e]     = ok ? lo[e] : (__bf16)0.0f;
    a[e + 8] = ok ? hh[e] : (__bf16)0.0f;
  }
  return a;
}

template<bool FULL>
__device__ inline v16bf load_Bt16(const __bf16* __restrict__ X, int ld, long rowBase,
                                  int nRows, int k0) {
  int lane = threadIdx.x & 31, n = lane & 15, hi = lane >> 4;
  int rc = FULL ? n : (n < nRows ? n : 0);
  v16bf b = *(const v16bf*)(X + (rowBase + rc) * (long)ld + k0 + hi * 16);
  if (!FULL) {
    bool ok = n < nRows;
#pragma unroll
    for (int e = 0; e < 16; ++e) b[e] = ok ? b[e] : (__bf16)0.0f;
  }
  return b;
}

// ---------------------------------------------------------------------------
// Weight pack: fp32 [K,Nc] -> bf16 tile-major [kt][nt][lane][16], zero-padded to
// Kp x Ncp. GEMM lanes then read one contiguous 32B block each.
// ---------------------------------------------------------------------------
__global__ void pack_B_kernel(const float* __restrict__ W, int K, int Nc,
                              __bf16* __restrict__ Bp, int Kp, int Ncp) {
  int idx = blockIdx.x * blockDim.x + threadIdx.x;
  int total = (Kp >> 5) * (Ncp >> 4) * 512;
  if (idx >= total) return;
  int e = idx & 15;
  int lane = (idx >> 4) & 31;
  int tile = idx >> 9;
  int ntn = Ncp >> 4;
  int kt = tile / ntn, nt = tile % ntn;
  int hi = lane >> 4, n = lane & 15;
  int k = kt * 32 + hi * 16 + e;
  int col = nt * 16 + n;
  float v = (k < K && col < Nc) ? W[(long)k * Nc + col] : 0.0f;
  Bp[idx] = (__bf16)v;
}

// ---------------------------------------------------------------------------
// GEMM: out = act(A[M,Kp] @ B + bias). A bf16 row-major (lda=Kp, rows mult of 16,
// K-pad zeroed), B packed. Each wave computes a 16 x (16*NT) strip.
// Software-pipelined: k-step (kt+1) tiles are prefetched before the kt WMMAs.
// outB (optional): bf16, ld=Ncp (pad cols get exact zeros).
// outF (optional): fp32, ld=Nc, guarded store.
// ---------------------------------------------------------------------------
template<int NT>
__global__ void wmma_gemm_kernel(const __bf16* __restrict__ A, int lda,
                                 const __bf16* __restrict__ Bp,
                                 const float* __restrict__ bias,
                                 float* outF, __bf16* outB,
                                 int M, int Kp, int Nc, int Ncp, int act) {
  int nTilesN = Ncp >> 4;
  int nGroups = nTilesN / NT;
  int tile = blockIdx.x;
  int tg = tile % nGroups;
  long rowBase = (long)(tile / nGroups) * 16;
  int tn0 = tg * NT;
  int lane = threadIdx.x & 31;

  v8f c[NT];
#pragma unroll
  for (int t = 0; t < NT; ++t) c[t] = (v8f){0.f,0.f,0.f,0.f,0.f,0.f,0.f,0.f};

  int kSteps = Kp >> 5;
  // prologue: load k-step 0
  v16bf a = load_A16<true>(A, lda, rowBase, 16, 0);
  v16bf b[NT];
#pragma unroll
  for (int t = 0; t < NT; ++t)
    b[t] = *(const v16bf*)(Bp + ((long)(tn0 + t)) * 512 + (long)lane * 16);

  for (int kt = 0; kt < kSteps - 1; ++kt) {
    // prefetch k-step kt+1 while the kt WMMAs execute
    v16bf an = load_A16<true>(A, lda, rowBase, 16, (kt + 1) * 32);
    v16bf bn[NT];
#pragma unroll
    for (int t = 0; t < NT; ++t)
      bn[t] = *(const v16bf*)(Bp + ((long)(kt + 1) * nTilesN + tn0 + t) * 512 +
                              (long)lane * 16);
#pragma unroll
    for (int t = 0; t < NT; ++t)
      c[t] = __builtin_amdgcn_wmma_f32_16x16x32_bf16(false, a, false, b[t], (short)0, c[t],
                                                     false, false);
    a = an;
#pragma unroll
    for (int t = 0; t < NT; ++t) b[t] = bn[t];
  }
#pragma unroll
  for (int t = 0; t < NT; ++t)
    c[t] = __builtin_amdgcn_wmma_f32_16x16x32_bf16(false, a, false, b[t], (short)0, c[t],
                                                   false, false);

  int rb = (lane >> 4) * 8;
#pragma unroll
  for (int t = 0; t < NT; ++t) {
    int col = (tn0 + t) * 16 + (lane & 15);
    int cc = col < Nc ? col : Nc - 1;
    float bv = bias[cc];
    bv = col < Nc ? bv : 0.0f;
#pragma unroll
    for (int v = 0; v < 8; ++v) {
      long row = rowBase + rb + v;
      float val = c[t][v] + bv;
      if (act == 1) {  // selu
        const float sc = 1.0507009873554805f, al = 1.6732632423543772f;
        val = val > 0.0f ? sc * val : sc * al * (__expf(val) - 1.0f);
      }
      if (outB) outB[row * Ncp + col] = (__bf16)val;
      if (outF && col < Nc) outF[row * Nc + col] = val;
    }
  }
}

// ---------------------------------------------------------------------------
// top-3 merge of two sorted triples (ascending dist, tie -> smaller index)
// ---------------------------------------------------------------------------
__device__ inline void top3_merge(float& a0, int& x0, float& a1, int& x1, float& a2, int& x2,
                                  float b0, int y0, float b1, int y1, float b2, int y2) {
#define LESS_(da, ja, db, jb) ((da) < (db) || ((da) == (db) && (ja) < (jb)))
  bool t0 = LESS_(a0, x0, b0, y0);
  float m0 = t0 ? a0 : b0; int n0 = t0 ? x0 : y0;
  float a0b = t0 ? a1 : a0; int x0b = t0 ? x1 : x0;
  float a1b = t0 ? a2 : a1; int x1b = t0 ? x2 : x1;
  float b0b = t0 ? b0 : b1; int y0b = t0 ? y0 : y1;
  float b1b = t0 ? b1 : b2; int y1b = t0 ? y1 : y2;
  bool t1 = LESS_(a0b, x0b, b0b, y0b);
  float m1 = t1 ? a0b : b0b; int n1 = t1 ? x0b : y0b;
  float a0c = t1 ? a1b : a0b; int x0c = t1 ? x1b : x0b;
  float b0c = t1 ? b0b : b1b; int y0c = t1 ? y0b : y1b;
  bool t2 = LESS_(a0c, x0c, b0c, y0c);
  float m2 = t2 ? a0c : b0c; int n2 = t2 ? x0c : y0c;
  a0 = m0; x0 = n0; a1 = m1; x1 = n1; a2 = m2; x2 = n2;
#undef LESS_
}

// ---------------------------------------------------------------------------
// Per-graph kNN(k=3) via WMMA dot tiles. One wave per 16-row i-tile; the i-tile
// operand (B) is loop-invariant and hoisted; A streams the j tiles with a
// one-tile-deep software pipeline. Lane l handles i = iBase+(l&15); lane^16
// holds the other j half; merged via shfl at the end.
// ---------------------------------------------------------------------------
template<int C, bool FULL>
__global__ void wmma_knn_kernel(const __bf16* __restrict__ X, const float* __restrict__ sq,
                                int nPer, int* __restrict__ knn_out) {
  constexpr int KS = C / 32;
  int tilesPerGraph = (nPer + 15) >> 4;
  int g = blockIdx.x / tilesPerGraph;
  int it = blockIdx.x % tilesPerGraph;
  long gOff = (long)g * nPer;
  int iBase = it * 16;
  int lane = threadIdx.x & 31;
  int iLoc = iBase + (lane & 15);
  int hi = lane >> 4;
  int nI = nPer - iBase; if (nI > 16) nI = 16;

  v16bf bt[KS];
#pragma unroll
  for (int ks = 0; ks < KS; ++ks)
    bt[ks] = load_Bt16<FULL>(X, C, gOff + iBase, nI, ks * 32);

  float d0 = 1e30f, d1 = 1e30f, d2 = 1e30f;
  int j0 = 0x7fffffff, j1 = 0x7fffffff, j2 = 0x7fffffff;
  int iC = iLoc < nPer ? iLoc : 0;
  float sqi = sq[gOff + iC];

  // pipeline prologue: tile jBase = 0
  int nJ0 = nPer < 16 ? nPer : 16;
  v16bf aj[KS];
#pragma unroll
  for (int ks = 0; ks < KS; ++ks)
    aj[ks] = load_A16<FULL>(X, C, gOff, nJ0, ks * 32);
  v8f sqj;
  if (FULL) sqj = *(const v8f*)(sq + gOff + hi * 8);

  for (int jBase = 0; jBase < nPer; jBase += 16) {
    // prefetch next tile (clamped to 0 on last iteration; harmless re-read)
    int jn = (jBase + 16 < nPer) ? jBase + 16 : 0;
    int nJn = nPer - jn; if (nJn > 16) nJn = 16;
    v16bf an[KS];
#pragma unroll
    for (int ks = 0; ks < KS; ++ks)
      an[ks] = load_A16<FULL>(X, C, gOff + jn, nJn, ks * 32);
    v8f sqn;
    if (FULL) sqn = *(const v8f*)(sq + gOff + jn + hi * 8);

    v8f c = {0.f,0.f,0.f,0.f,0.f,0.f,0.f,0.f};
#pragma unroll
    for (int ks = 0; ks < KS; ++ks)
      c = __builtin_amdgcn_wmma_f32_16x16x32_bf16(false, aj[ks], false, bt[ks], (short)0, c,
                                                  false, false);
#pragma unroll
    for (int v = 0; v < 8; ++v) {
      int j = jBase + hi * 8 + v;
      float sqv;
      if (FULL) {
        sqv = sqj[v];
      } else {
        int jc = j < nPer ? j : 0;
        sqv = sq[gOff + jc];
      }
      float d = sqi + sqv - 2.0f * c[v];
      bool valid = (FULL || j < nPer) && (j != iLoc);
      if (valid) {
        if (d < d0 || (d == d0 && j < j0)) {
          d2 = d1; j2 = j1; d1 = d0; j1 = j0; d0 = d; j0 = j;
        } else if (d < d1 || (d == d1 && j < j1)) {
          d2 = d1; j2 = j1; d1 = d; j1 = j;
        } else if (d < d2 || (d == d2 && j < j2)) {
          d2 = d; j2 = j;
        }
      }
    }
#pragma unroll
    for (int ks = 0; ks < KS; ++ks) aj[ks] = an[ks];
    if (FULL) sqj = sqn;
  }

  float pd0 = __shfl_xor(d0, 16, 32), pd1 = __shfl_xor(d1, 16, 32), pd2 = __shfl_xor(d2, 16, 32);
  int pj0 = __shfl_xor(j0, 16, 32), pj1 = __shfl_xor(j1, 16, 32), pj2 = __shfl_xor(j2, 16, 32);
  top3_merge(d0, j0, d1, j1, d2, j2, pd0, pj0, pd1, pj1, pd2, pj2);

  if (hi == 0 && iLoc < nPer) {
    long node = gOff + iLoc;
    knn_out[node * 3 + 0] = (int)gOff + j0;
    knn_out[node * 3 + 1] = (int)gOff + j1;
    knn_out[node * 3 + 2] = (int)gOff + j2;
  }
}

// ---------------------------------------------------------------------------
// Elementwise / gather / reduction helper kernels (bandwidth-bound)
// ---------------------------------------------------------------------------
__global__ void prep_kernel(const float* __restrict__ X, int n, int C,
                            __bf16* __restrict__ Xb, float* __restrict__ sq) {
  int i = blockIdx.x * blockDim.x + threadIdx.x;
  if (i >= n) return;
  float s = 0.0f;
  for (int c = 0; c < C; ++c) {
    float v = X[(long)i * C + c];
    s += v * v;
    Xb[(long)i * C + c] = (__bf16)v;
  }
  sq[i] = s;
}

__global__ void f32_to_bf16_slice(const float* __restrict__ X, int n, int C,
                                  __bf16* __restrict__ out, int ldo, int co) {
  int idx = blockIdx.x * blockDim.x + threadIdx.x;
  if (idx >= n * C) return;
  int i = idx / C, c = idx % C;
  out[(long)i * ldo + co + c] = (__bf16)X[idx];
}

__global__ void dn_kernel(const float* __restrict__ w, float* __restrict__ dn, int n) {
  int i = blockIdx.x * blockDim.x + threadIdx.x;
  if (i >= n) return;
  float deg = w[3 * i] + w[3 * i + 1] + w[3 * i + 2];
  dn[i] = deg > 0.0f ? rsqrtf(fmaxf(deg, 1e-12f)) : 0.0f;
}

// r = -alpha * segsum(wn * v[src]) + beta * prev ; writes fp32 (optional) + bf16 slice
__global__ void lap_kernel(const float* __restrict__ v, const float* __restrict__ prev,
                           const float* __restrict__ w, const int* __restrict__ src,
                           const float* __restrict__ dn, int n, int C,
                           float alpha, float beta,
                           float* outF, __bf16* __restrict__ outB, int ldo, int co) {
  int idx = blockIdx.x * blockDim.x + threadIdx.x;
  if (idx >= n * C) return;
  int i = idx / C, c = idx % C;
  float di = dn[i];
  int s0 = src[3 * i], s1 = src[3 * i + 1], s2 = src[3 * i + 2];
  float l = w[3 * i]     * di * dn[s0] * v[(long)s0 * C + c]
          + w[3 * i + 1] * di * dn[s1] * v[(long)s1 * C + c]
          + w[3 * i + 2] * di * dn[s2] * v[(long)s2 * C + c];
  float r = -alpha * l + (prev ? beta * prev[idx] : 0.0f);
  if (outF) outF[idx] = r;
  outB[(long)i * ldo + co + c] = (__bf16)r;
}

__global__ void gn_stats_kernel(const float* __restrict__ X, const float* __restrict__ alpha,
                                float* __restrict__ mean, float* __restrict__ var,
                                int nPer, int C) {
  int b = blockIdx.x / C, c = blockIdx.x % C;
  __shared__ float s1[256], s2[256];
  float a1 = 0.0f, a2 = 0.0f;
  for (int n = threadIdx.x; n < nPer; n += 256) {
    float v = X[((long)b * nPer + n) * C + c];
    a1 += v; a2 += v * v;
  }
  s1[threadIdx.x] = a1; s2[threadIdx.x] = a2;
  __syncthreads();
  for (int off = 128; off > 0; off >>= 1) {
    if (threadIdx.x < off) { s1[threadIdx.x] += s1[threadIdx.x + off];
                             s2[threadIdx.x] += s2[threadIdx.x + off]; }
    __syncthreads();
  }
  if (threadIdx.x == 0) {
    float mu = s1[0] / nPer;
    float al = alpha[c];
    var[blockIdx.x] = s2[0] / nPer - 2.0f * al * mu * mu + al * al * mu * mu;
    mean[blockIdx.x] = mu;
  }
}

__global__ void gn_apply_kernel(const float* __restrict__ X, const float* __restrict__ mean,
                                const float* __restrict__ var, const float* __restrict__ alpha,
                                const float* __restrict__ gamma, const float* __restrict__ beta,
                                float* __restrict__ out, int nPer, int C, int nTot) {
  int idx = blockIdx.x * blockDim.x + threadIdx.x;
  if (idx >= nTot * C) return;
  int i = idx / C, c = idx % C;
  int b = i / nPer;
  float sub = X[idx] - alpha[c] * mean[b * C + c];
  out[idx] = gamma[c] * sub * rsqrtf(var[b * C + c] + 1e-5f) + beta[c];
}

__global__ void softmax_prep_kernel(const float* __restrict__ U, int n, int m,
                                    float* __restrict__ mx, float* __restrict__ den) {
  int i = blockIdx.x * blockDim.x + threadIdx.x;
  if (i >= n) return;
  float mm = -1e30f;
  for (int j = 0; j < m; ++j) mm = fmaxf(mm, U[(long)i * m + j]);
  float s = 0.0f;
  for (int j = 0; j < m; ++j) s += __expf(U[(long)i * m + j] - mm);
  mx[i] = mm; den[i] = s;
}

// DiffPool: pooled[b,m,c] = sum_n S[b,n,m] X[b,n,c] / sum_n S[b,n,m]
__global__ void pool_kernel(const float* __restrict__ U, const float* __restrict__ mx,
                            const float* __restrict__ den, const float* __restrict__ X,
                            float* __restrict__ xp) {
  int b = blockIdx.x / MP_, m = blockIdx.x % MP_;
  int t = threadIdx.x;
  int c = t & 31, slice = t >> 5;  // 8 slices x 32 channels
  __shared__ float s_sh[256];
  __shared__ float red[256];
  __shared__ float ss[8];
  __shared__ float ssum_tot;
  float acc = 0.0f, ssum = 0.0f;
  for (int base = 0; base < NG_; base += 256) {
    long gi = (long)b * NG_ + base + t;
    s_sh[t] = __expf(U[gi * MP_ + m] - mx[gi]) / den[gi];
    __syncthreads();
    for (int k = slice; k < 256; k += 8) {
      float sv = s_sh[k];
      acc += sv * X[((long)b * NG_ + base + k) * IN_CH_ + c];
      if (c == 0) ssum += sv;
    }
    __syncthreads();
  }
  red[t] = acc;
  if (c == 0) ss[slice] = ssum;
  __syncthreads();
  if (t < 128) red[t] += red[t + 128];
  __syncthreads();
  if (t < 64) red[t] += red[t + 64];
  __syncthreads();
  if (t == 0) {
    float s = 0.0f;
    for (int k = 0; k < 8; ++k) s += ss[k];
    ssum_tot = s;
  }
  __syncthreads();
  if (t < 32) {
    float tot = red[t] + red[t + 32];
    xp[((long)b * MP_ + m) * IN_CH_ + t] = tot / ssum_tot;
  }
}

__global__ void zero_u32_kernel(unsigned* p, int n) {
  int i = blockIdx.x * blockDim.x + threadIdx.x;
  if (i < n) p[i] = 0u;
}

// EdgeConv over fixed PPI graph; relu>=0 so uint atomicMax on float bits == float max
__global__ void edgeconv_ppi_kernel(const int* __restrict__ connect, const float* __restrict__ xp,
                                    const float* __restrict__ W, const float* __restrict__ bias,
                                    unsigned* __restrict__ agg) {
  int e = blockIdx.x;
  int bg = e / E_PPI_, le = e % E_PPI_;
  int s = connect[le] + bg * MP_;
  int d = connect[E_PPI_ + le] + bg * MP_;
  __shared__ float feat[2 * IN_CH_];
  int t = threadIdx.x;
  if (t < IN_CH_) {
    float xd = xp[(long)d * IN_CH_ + t];
    float xs = xp[(long)s * IN_CH_ + t];
    feat[t] = xd;
    feat[IN_CH_ + t] = xs - xd;
  }
  __syncthreads();
  float a = bias[t];
  for (int k = 0; k < 2 * IN_CH_; ++k) a += feat[k] * W[k * HID_ + t];
  a = fmaxf(a, 0.0f);
  atomicMax(&agg[(long)d * HID_ + t], __float_as_uint(a));
}

__global__ void edgeconv_dyn_kernel(const int* __restrict__ knn, const float* __restrict__ h,
                                    const float* __restrict__ W, const float* __restrict__ bias,
                                    unsigned* __restrict__ agg) {
  int e = blockIdx.x;
  int d = e / 3;
  int s = knn[e];
  __shared__ float feat[2 * HID_];
  int t = threadIdx.x;
  float hd = h[(long)d * HID_ + t];
  float hs = h[(long)s * HID_ + t];
  feat[t] = hd;
  feat[HID_ + t] = hs - hd;
  __syncthreads();
  float a = bias[t];
  for (int k = 0; k < 2 * HID_; ++k) a += feat[k] * W[k * HID_ + t];
  a = fmaxf(a, 0.0f);
  atomicMax(&agg[(long)d * HID_ + t], __float_as_uint(a));
}

__global__ void add_f32_kernel(const unsigned* __restrict__ a, const unsigned* __restrict__ b,
                               float* __restrict__ out, int n) {
  int i = blockIdx.x * blockDim.x + threadIdx.x;
  if (i < n) out[i] = __uint_as_float(a[i]) + __uint_as_float(b[i]);
}

__global__ void small_gemm_kernel(const float* __restrict__ A, int M, int K,
                                  const float* __restrict__ W, const float* __restrict__ bias,
                                  int Nc, float* __restrict__ out, int act) {
  int idx = blockIdx.x * blockDim.x + threadIdx.x;
  if (idx >= M * Nc) return;
  int i = idx / Nc, j = idx % Nc;
  float a = bias[j];
  for (int k = 0; k < K; ++k) a += A[(long)i * K + k] * W[k * Nc + j];
  if (act == 1) {
    const float sc = 1.0507009873554805f, al = 1.6732632423543772f;
    a = a > 0.0f ? sc * a : sc * al * (__expf(a) - 1.0f);
  }
  out[idx] = a;
}

// ---------------------------------------------------------------------------
static inline int cdiv(int a, int b) { return (a + b - 1) / b; }
static inline int rup32(int a) { return (a + 31) & ~31; }

static void launch_gemm(const __bf16* A, int lda, const __bf16* Bp, const float* bias,
                        float* outF, __bf16* outB, int M, int Kp, int Nc, int Ncp, int act,
                        hipStream_t s) {
  int tilesM = M / 16, tilesN = Ncp / 16;
  if (tilesN % 4 == 0)
    wmma_gemm_kernel<4><<<tilesM * (tilesN / 4), 32, 0, s>>>(A, lda, Bp, bias, outF, outB,
                                                             M, Kp, Nc, Ncp, act);
  else if (tilesN % 2 == 0)
    wmma_gemm_kernel<2><<<tilesM * (tilesN / 2), 32, 0, s>>>(A, lda, Bp, bias, outF, outB,
                                                             M, Kp, Nc, Ncp, act);
  else
    wmma_gemm_kernel<1><<<tilesM * tilesN, 32, 0, s>>>(A, lda, Bp, bias, outF, outB,
                                                       M, Kp, Nc, Ncp, act);
}

static void launch_pack(const float* W, int K, int Nc, __bf16* Bp, int Kp, int Ncp,
                        hipStream_t s) {
  int total = (Kp >> 5) * (Ncp >> 4) * 512;
  pack_B_kernel<<<cdiv(total, 256), 256, 0, s>>>(W, K, Nc, Bp, Kp, Ncp);
}

extern "C" void kernel_launch(void* const* d_in, const int* in_sizes, int n_in,
                              void* d_out, int out_size, void* d_ws, size_t ws_size,
                              hipStream_t stream) {
  const float* x    = (const float*)d_in[0];
  const float* ew   = (const float*)d_in[1];
  // d_in[2] = batch (unused)
  const int*   conn = (const int*)d_in[3];
  const float* W0   = (const float*)d_in[4];
  const float* b0   = (const float*)d_in[5];
  const float* gna  = (const float*)d_in[6];
  const float* gng  = (const float*)d_in[7];
  const float* gnb  = (const float*)d_in[8];
  const float* W1   = (const float*)d_in[9];
  const float* b1c  = (const float*)d_in[10];
  const float* f1w1 = (const float*)d_in[11];
  const float* f1b1 = (const float*)d_in[12];
  const float* f1w2 = (const float*)d_in[13];
  const float* f1b2 = (const float*)d_in[14];
  const float* f1w3 = (const float*)d_in[15];
  const float* f1b3 = (const float*)d_in[16];
  const float* W2   = (const float*)d_in[17];
  const float* b2   = (const float*)d_in[18];
  const float* W3   = (const float*)d_in[19];
  const float* b3   = (const float*)d_in[20];
  const float* f2w1 = (const float*)d_in[21];
  const float* f2b1 = (const float*)d_in[22];
  const float* f2w2 = (const float*)d_in[23];
  const float* f2b2 = (const float*)d_in[24];
  const float* f2w3 = (const float*)d_in[25];
  const float* f2b3 = (const float*)d_in[26];

  float* out_u = (float*)d_out;                 // [NTOT, 100]
  float* out_v = out_u + (long)NTOT * MP_;      // [800, 16]

  // padded leading dims (multiples of 32 so GEMM A-loads need no K guard)
  const int K0p  = 4 * IN_CH_;          // 128
  const int K1p  = 4 * HID_;            // 256
  const int NC1  = 4 * MP_;             // 400
  const int NC1p = rup32(NC1);          // 416
  const int NH1  = 2 * MP_;             // 200
  const int NH1p = rup32(NH1);          // 224
  const int NH2p = rup32(MP_);          // 128

  // ---- workspace carve ----
  char* ws = (char*)d_ws;
  auto alloc = [&](size_t bytes) -> void* {
    void* p = (void*)ws;
    ws += (bytes + 255) & ~(size_t)255;
    return p;
  };
  __bf16* xbf    = (__bf16*)alloc((size_t)NTOT * IN_CH_ * 2);
  float*  sqx    = (float*) alloc((size_t)NTOT * 4);
  int*    knn1   = (int*)   alloc((size_t)NTOT * 3 * 4);
  float*  dn     = (float*) alloc((size_t)NTOT * 4);
  float*  Z1f    = (float*) alloc((size_t)NTOT * HID_ * 4);
  float*  Z2f    = (float*) alloc((size_t)NTOT * HID_ * 4);
  __bf16* Zcat0  = (__bf16*)alloc((size_t)NTOT * K0p * 2);
  float*  u0     = (float*) alloc((size_t)NTOT * HID_ * 4);
  float*  gmean  = (float*) alloc((size_t)B_GR * HID_ * 4);
  float*  gvar   = (float*) alloc((size_t)B_GR * HID_ * 4);
  float*  un     = (float*) alloc((size_t)NTOT * HID_ * 4);
  __bf16* ubf    = (__bf16*)alloc((size_t)NTOT * HID_ * 2);
  float*  squ    = (float*) alloc((size_t)NTOT * 4);
  __bf16* Zcat1  = (__bf16*)alloc((size_t)NTOT * K1p * 2);
  __bf16* c1bf   = (__bf16*)alloc((size_t)NTOT * NC1p * 2);
  __bf16* h1bf   = (__bf16*)alloc((size_t)NTOT * NH1p * 2);
  __bf16* h2bf   = (__bf16*)alloc((size_t)NTOT * NH2p * 2);
  __bf16* Bp0    = (__bf16*)alloc((size_t)(K0p/32) * (HID_/16) * 512 * 2);
  __bf16* Bp1    = (__bf16*)alloc((size_t)(K1p/32) * (NC1p/16) * 512 * 2);
  __bf16* Bpf1   = (__bf16*)alloc((size_t)(NC1p/32) * (NH1p/16) * 512 * 2);
  __bf16* Bpf2   = (__bf16*)alloc((size_t)(NH1p/32) * (NH2p/16) * 512 * 2);
  __bf16* Bpf3   = (__bf16*)alloc((size_t)(NH2p/32) * (NH2p/16) * 512 * 2);
  float*  mxb    = (float*) alloc((size_t)NTOT * 4);
  float*  denb   = (float*) alloc((size_t)NTOT * 4);
  float*  xp     = (float*) alloc((size_t)NPOOL * IN_CH_ * 4);
  unsigned* hppi = (unsigned*)alloc((size_t)NPOOL * HID_ * 4);
  __bf16* hbf    = (__bf16*)alloc((size_t)NPOOL * HID_ * 2);
  float*  sqh    = (float*) alloc((size_t)NPOOL * 4);
  int*    knn2   = (int*)   alloc((size_t)NPOOL * 3 * 4);
  unsigned* yagg = (unsigned*)alloc((size_t)NPOOL * HID_ * 4);
  float*  zbuf   = (float*) alloc((size_t)NPOOL * HID_ * 4);
  float*  g1     = (float*) alloc((size_t)NPOOL * (HID_ / 2) * 4);
  float*  g2     = (float*) alloc((size_t)NPOOL * (HID_ / 4) * 4);

  // ---- weight packs (tiny) ----
  launch_pack(W0,   K0p,  HID_, Bp0,  K0p,  HID_, stream);
  launch_pack(W1,   K1p,  NC1,  Bp1,  K1p,  NC1p, stream);
  launch_pack(f1w1, NC1,  NH1,  Bpf1, NC1p, NH1p, stream);
  launch_pack(f1w2, NH1,  MP_,  Bpf2, NH1p, NH2p, stream);
  launch_pack(f1w3, MP_,  MP_,  Bpf3, NH2p, NH2p, stream);

  // ---- stage 1: kNN on raw x (C=32) ----
  prep_kernel<<<cdiv(NTOT, 256), 256, 0, stream>>>(x, NTOT, IN_CH_, xbf, sqx);
  wmma_knn_kernel<IN_CH_, true><<<B_GR * (NG_ / 16), 32, 0, stream>>>(xbf, sqx, NG_, knn1);
  dn_kernel<<<cdiv(NTOT, 256), 256, 0, stream>>>(ew, dn, NTOT);

  // ---- cheb conv0: Zcat0 = [x, Z1, Z2, Z3] bf16 ; u0 = Zcat0 @ W0cat + b0 ----
  {
    int nc = NTOT * IN_CH_;
    f32_to_bf16_slice<<<cdiv(nc, 256), 256, 0, stream>>>(x, NTOT, IN_CH_, Zcat0, K0p, 0);
    lap_kernel<<<cdiv(nc, 256), 256, 0, stream>>>(x, nullptr, ew, knn1, dn, NTOT, IN_CH_,
                                                  1.0f, 0.0f, Z1f, Zcat0, K0p, IN_CH_);
    lap_kernel<<<cdiv(nc, 256), 256, 0, stream>>>(Z1f, x, ew, knn1, dn, NTOT, IN_CH_,
                                                  2.0f, -1.0f, Z2f, Zcat0, K0p, 2 * IN_CH_);
    lap_kernel<<<cdiv(nc, 256), 256, 0, stream>>>(Z2f, Z1f, ew, knn1, dn, NTOT, IN_CH_,
                                                  2.0f, -1.0f, nullptr, Zcat0, K0p, 3 * IN_CH_);
    launch_gemm(Zcat0, K0p, Bp0, b0, u0, nullptr, NTOT, K0p, HID_, HID_, 0, stream);
  }

  // ---- GraphNorm ----
  gn_stats_kernel<<<B_GR * HID_, 256, 0, stream>>>(u0, gna, gmean, gvar, NG_, HID_);
  gn_apply_kernel<<<cdiv(NTOT * HID_, 256), 256, 0, stream>>>(u0, gmean, gvar, gna, gng, gnb,
                                                              un, NG_, HID_, NTOT);

  // ---- kNN on hidden (C=64) ----
  prep_kernel<<<cdiv(NTOT, 256), 256, 0, stream>>>(un, NTOT, HID_, ubf, squ);
  wmma_knn_kernel<HID_, true><<<B_GR * (NG_ / 16), 32, 0, stream>>>(ubf, squ, NG_, knn1);

  // ---- cheb conv1: Zcat1[N,256] @ W1cat[256,400->416] -> c1bf ----
  {
    int nc = NTOT * HID_;
    f32_to_bf16_slice<<<cdiv(nc, 256), 256, 0, stream>>>(un, NTOT, HID_, Zcat1, K1p, 0);
    lap_kernel<<<cdiv(nc, 256), 256, 0, stream>>>(un, nullptr, ew, knn1, dn, NTOT, HID_,
                                                  1.0f, 0.0f, Z1f, Zcat1, K1p, HID_);
    lap_kernel<<<cdiv(nc, 256), 256, 0, stream>>>(Z1f, un, ew, knn1, dn, NTOT, HID_,
                                                  2.0f, -1.0f, Z2f, Zcat1, K1p, 2 * HID_);
    lap_kernel<<<cdiv(nc, 256), 256, 0, stream>>>(Z2f, Z1f, ew, knn1, dn, NTOT, HID_,
                                                  2.0f, -1.0f, nullptr, Zcat1, K1p, 3 * HID_);
    launch_gemm(Zcat1, K1p, Bp1, b1c, nullptr, c1bf, NTOT, K1p, NC1, NC1p, 0, stream);
  }

  // ---- fc1 MLP: 400 -> 200 (selu) -> 100 (selu) -> 100 ; final into d_out (u) ----
  launch_gemm(c1bf, NC1p, Bpf1, f1b1, nullptr, h1bf, NTOT, NC1p, NH1, NH1p, 1, stream);
  launch_gemm(h1bf, NH1p, Bpf2, f1b2, nullptr, h2bf, NTOT, NH1p, MP_, NH2p, 1, stream);
  launch_gemm(h2bf, NH2p, Bpf3, f1b3, out_u, nullptr, NTOT, NH2p, MP_, NH2p, 0, stream);

  // ---- DiffPool soft assignment ----
  softmax_prep_kernel<<<cdiv(NTOT, 256), 256, 0, stream>>>(out_u, NTOT, MP_, mxb, denb);
  pool_kernel<<<B_GR * MP_, 256, 0, stream>>>(out_u, mxb, denb, x, xp);

  // ---- PPI EdgeConv ----
  zero_u32_kernel<<<cdiv(NPOOL * HID_, 256), 256, 0, stream>>>(hppi, NPOOL * HID_);
  edgeconv_ppi_kernel<<<B_GR * E_PPI_, HID_, 0, stream>>>(conn, xp, W2, b2, hppi);

  // ---- Dynamic EdgeConv: kNN(k=3) on h (100 nodes/graph, C=64) ----
  prep_kernel<<<cdiv(NPOOL, 256), 256, 0, stream>>>((const float*)hppi, NPOOL, HID_, hbf, sqh);
  wmma_knn_kernel<HID_, false><<<B_GR * cdiv(MP_, 16), 32, 0, stream>>>(hbf, sqh, MP_, knn2);
  zero_u32_kernel<<<cdiv(NPOOL * HID_, 256), 256, 0, stream>>>(yagg, NPOOL * HID_);
  edgeconv_dyn_kernel<<<NPOOL * 3, HID_, 0, stream>>>(knn2, (const float*)hppi, W3, b3, yagg);
  add_f32_kernel<<<cdiv(NPOOL * HID_, 256), 256, 0, stream>>>(hppi, yagg, zbuf, NPOOL * HID_);

  // ---- fc2 MLP: 64 -> 32 (selu) -> 16 (selu) -> 16 ; final into d_out (v) ----
  small_gemm_kernel<<<cdiv(NPOOL * 32, 256), 256, 0, stream>>>(zbuf, NPOOL, HID_, f2w1, f2b1, 32, g1, 1);
  small_gemm_kernel<<<cdiv(NPOOL * 16, 256), 256, 0, stream>>>(g1, NPOOL, 32, f2w2, f2b2, 16, g2, 1);
  small_gemm_kernel<<<cdiv(NPOOL * 16, 256), 256, 0, stream>>>(g2, NPOOL, 16, f2w3, f2b3, 16, out_v, 0);

  (void)in_sizes; (void)n_in; (void)out_size; (void)ws_size;
}
